// RingDilatedAttentionRefactored_68959994905212
// MI455X (gfx1250) — matrix operand order repaired and checked by reading, not compiled, not run on hardware
//
#include <hip/hip_runtime.h>
#include <hip/hip_bf16.h>
#include <math.h>

// ---------------------------------------------------------------------------
// RingDilatedAttention for MI455X (gfx1250, wave32).
//   B=1, N=32768, H=16, D=64, segments=[128,128], dilations=[1,2].
// Only rows 0..255 of the output are nonzero; the 134 MB output zero-fill is
// the bandwidth floor (~6us @ 23.3 TB/s). K/V gathers add ~268 MB. Compute is
// ~0.27 GFLOP -> irrelevant; P*V is mapped to v_wmma_f32_16x16x32_f16 via a
// diagonal-A construction (key sets are per-query, so plain GEMM can't apply).
//
// All pattern positions are provably < N for this config:
//   seg0: max = 127 + 128*127 = 16383;  seg1: max = 127 + 256*127 = 32639.
// So global loads are unconditional (no EXEC-diverged per-element guards),
// staged into register arrays and double-buffered so each inner iteration
// pays one s_wait that overlaps with the WMMA instead of 16 serial waits.
// ---------------------------------------------------------------------------

typedef __attribute__((ext_vector_type(16))) _Float16 v16h;
typedef __attribute__((ext_vector_type(8)))  float    v8f;
typedef __attribute__((ext_vector_type(4)))  float    f4;

#define N_SEQ   32768
#define NHEAD   16
#define DHEAD   64
#define HD      (NHEAD * DHEAD)   // 1024 floats per sequence position
#define SEG_LEN 128
#define NEG_INF (-1e30f)
#define SCALE   0.125f            // 1/sqrt(64), exact

// ---------------- Kernel 1: zero the whole output (bandwidth floor) --------
__global__ __launch_bounds__(256) void dila_zero_kernel(f4* __restrict__ out, long n4) {
    long i      = (long)blockIdx.x * blockDim.x + threadIdx.x;
    long stride = (long)gridDim.x * blockDim.x;
    f4 z = (f4)0.0f;
    for (; i < n4; i += stride)
        __builtin_nontemporal_store(z, &out[i]);   // NT: don't pollute L2 with zeros
}

// ---------------- Kernel 2: attention over the 2x128 active queries --------
// grid = (qtile=8, head=16, seg=2), block = 128 threads (4 wave32).
__global__ __launch_bounds__(128) void dila_attn_kernel(
    const float* __restrict__ q, const float* __restrict__ k,
    const float* __restrict__ v, const int* __restrict__ is_causal_p,
    float* __restrict__ out)
{
    const int qtile     = blockIdx.x;               // 16 queries per block
    const int h         = blockIdx.y;
    const int seg       = blockIdx.z;
    const int seg_start = seg * SEG_LEN;            // 0 or 128
    const int step      = (seg == 0) ? 128 : 256;   // dilation * seg_len
    const int i0        = qtile * 16;               // local query base in segment
    const int t         = threadIdx.x;
    const int causal    = is_causal_p[0];

    __shared__ float    sS[16][128];                // fp32 scores
    __shared__ _Float16 sP[16][128];                // f16 probabilities for WMMA

    // ---- Phase 1: scores. 8 threads per query; thread owns an 8-dim slice.
    // Per key column the block touches 16 consecutive sequence positions
    // (i0+j+step*kk, j=0..15) -> coalesced 2x128B chunks per K row.
    // Double-buffered so the next K slice is in flight during the reduction.
    {
        const int j = t >> 3, c = t & 7;
        const int qpos = seg_start + i0 + j;
        const float* qrow = q + (size_t)qpos * HD + h * DHEAD;
        const f4 q0 = *(const f4*)(qrow + c * 4);
        const f4 q1 = *(const f4*)(qrow + 32 + c * 4);

        const float* krow0 = k + (size_t)(i0 + j) * HD + h * DHEAD;
        f4 kc0 = *(const f4*)(krow0 + c * 4);
        f4 kc1 = *(const f4*)(krow0 + 32 + c * 4);

        for (int kk = 0; kk < 128; ++kk) {
            f4 kn0, kn1;
            if (kk < 127) {                         // prefetch next key row slice
                const float* krow =
                    k + (size_t)(i0 + j + step * (kk + 1)) * HD + h * DHEAD;
                kn0 = *(const f4*)(krow + c * 4);
                kn1 = *(const f4*)(krow + 32 + c * 4);
            }
            float part = q0.x*kc0.x + q0.y*kc0.y + q0.z*kc0.z + q0.w*kc0.w
                       + q1.x*kc1.x + q1.y*kc1.y + q1.z*kc1.z + q1.w*kc1.w;
            part += __shfl_xor(part, 1, 32);        // reduce across the 8-lane group
            part += __shfl_xor(part, 2, 32);
            part += __shfl_xor(part, 4, 32);
            if (c == 0) {
                float sc = part * SCALE;
                // causal: q_pos < key_pos  <=>  step*kk > seg_start
                if (causal && (step * kk > seg_start)) sc = NEG_INF;
                sS[j][kk] = sc;
            }
            if (kk < 127) { kc0 = kn0; kc1 = kn1; }
        }
    }
    __syncthreads();

    // ---- Phase 2: fp32 softmax per query (128 entries), emit f16 P.
    if (t < 16) {
        float m = -3.402823e38f;
        for (int kk = 0; kk < 128; ++kk) m = fmaxf(m, sS[t][kk]);
        float ssum = 0.0f;
        for (int kk = 0; kk < 128; ++kk) {
            float e = expf(sS[t][kk] - m);
            sS[t][kk] = e;
            ssum += e;
        }
        float inv = 1.0f / ssum;
        for (int kk = 0; kk < 128; ++kk)
            sP[t][kk] = (_Float16)(sS[t][kk] * inv);
    }
    __syncthreads();

    // ---- Phase 3: P*V via v_wmma_f32_16x16x32_f16 (diagonal-A trick).
    // Wave w handles output dims [16w, 16w+16). Per WMMA: 2 key columns.
    //   A[m, m]    = p[m, kc0]     A[m, 16+m] = p[m, kc1]   (else 0)
    //   B[K,  n]   = V[i0+K    + step*kc0][d0+n]  (K <16)
    //   B[16+e, n] = V[i0+e    + step*kc1][d0+n]
    // A layout (contiguous, matching B): lanes 0-15 row M=lane hold K=e;
    // lanes 16-31 row M=lane-16 hold K=16+e.
    // -> lane L<16 sets a[L]=p0; lane L>=16 sets a[L-16]=p1.
    // B layout used here (contiguous, matching A): lanes 0-15 col N=lane hold K=e; lanes 16-31 K=16+e.
    {
        const int lane = t & 31;
        const int wave = t >> 5;
        const int d0   = wave * 16;
        const int n    = lane & 15;
        const int e0   = (lane < 16) ? lane : -1;
        const int e1   = (lane < 16) ? -1 : (lane - 16);
        const int mr2  = lane & 15;
        const int kcof = (lane < 16) ? 0 : 1;   // which key column this lane's B holds

        v8f acc = {};
        const float* vbase = v + (size_t)h * DHEAD + d0 + n;

        // Stage 16 V elements (one B fragment) in fp32 registers; double-buffer.
        float cur[16];
        {
            const int pb = i0 + step * kcof;
            #pragma unroll
            for (int e = 0; e < 16; ++e)
                cur[e] = vbase[(size_t)(pb + e) * HD];
        }

        auto process = [&](const float* bf, int kc0) {
            const _Float16 p0 = sP[mr2][kc0];       // adjacent -> single ds_load_b32
            const _Float16 p1 = sP[mr2][kc0 + 1];
            v16h a;
            #pragma unroll
            for (int e = 0; e < 16; ++e)
                a[e] = (e == e0) ? p0 : ((e == e1) ? p1 : (_Float16)0);
            v16h b;
            #pragma unroll
            for (int e = 0; e < 16; ++e)
                b[e] = (_Float16)bf[e];
            acc = __builtin_amdgcn_wmma_f32_16x16x32_f16(
                      /*neg_a=*/false, a, /*neg_b=*/false, b,
                      /*c_mod=*/(short)0, acc, /*reuse_a=*/false, /*reuse_b=*/false);
        };

        for (int tp = 0; tp < 63; ++tp) {
            float nxt[16];
            const int pbn = i0 + step * (2 * (tp + 1) + kcof);
            #pragma unroll
            for (int e = 0; e < 16; ++e)            // issue next tile's 16 loads
                nxt[e] = vbase[(size_t)(pbn + e) * HD];
            process(cur, 2 * tp);                   // overlaps with loads in flight
            #pragma unroll
            for (int e = 0; e < 16; ++e) cur[e] = nxt[e];
        }
        process(cur, 126);                          // epilogue (tp = 63)

        // D layout: VGPR r -> row M=r (lanes 0-15) / M=8+r (lanes 16-31), N=lane&15.
        #pragma unroll
        for (int r = 0; r < 8; ++r) {
            const int mr   = (lane < 16) ? r : (r + 8);
            const int orow = seg_start + i0 + mr;
            out[(size_t)orow * HD + h * DHEAD + d0 + n] = acc[r];
        }
    }
}

// ---------------------------------------------------------------------------
extern "C" void kernel_launch(void* const* d_in, const int* in_sizes, int n_in,
                              void* d_out, int out_size, void* d_ws, size_t ws_size,
                              hipStream_t stream) {
    const float* q  = (const float*)d_in[0];
    const float* k  = (const float*)d_in[1];
    const float* v  = (const float*)d_in[2];
    const int* is_c = (const int*)d_in[3];
    float* out      = (float*)d_out;

    // 1) zero the full output (rows >=256 stay zero; rows 0..255 overwritten).
    const long n4 = (long)out_size / 4;
    dila_zero_kernel<<<8192, 256, 0, stream>>>((f4*)out, n4);

    // 2) attention over the 256 active queries.
    dim3 grid(8, NHEAD, 2);
    dila_attn_kernel<<<grid, 128, 0, stream>>>(q, k, v, is_c, out);
}
